// MSN_13804024889363
// MI455X (gfx1250) — matrix operand order repaired and checked by reading, not compiled, not run on hardware
//
#include <hip/hip_runtime.h>

// ---------------------------------------------------------------------------
// MSN forward for MI455X (gfx1250, wave32).
//  * conv1x1 layers -> f16 WMMA GEMMs (v_wmma_f32_16x16x32_f16).
//    Activations stored point-major [N][Kpad] so BOTH GEMM tiles stage with
//    contiguous global->LDS async copies (global_load_async_to_lds_b128,
//    ASYNCcnt, double-buffered), and BatchNorm stats stay row reductions on
//    the f32 Z buffer.
//  * Each wave owns a 2x2 grid of 16x16 accumulators (32x32 out) -> 4
//    back-to-back v_wmma per k-step, 8 waves -> 64x128 block tile.
//  * BN pre-bias cancels analytically; gmax branch of the 1088-K concat GEMM
//    is rank-reduced to a per-(batch,row) column add (K: 1088 -> 64).
//  * MST: 64 workgroups, all-LDS Prim's. MDS: 1 WG/batch, density in LDS.
// Workspace ~200MB carved from d_ws.
// ---------------------------------------------------------------------------

#define ASYNC_STAGE 1   // 1: global_load_async_to_lds_b128 pipeline, 0: sync fallback

typedef __attribute__((ext_vector_type(16))) _Float16 v16h;
typedef __attribute__((ext_vector_type(8)))  _Float16 v8h;
typedef __attribute__((ext_vector_type(8)))  float    v8f;

static constexpr int   BB     = 4;
static constexpr int   NPTS   = 8192;
static constexpr int   NPRIM  = 16;
static constexpr int   SS     = NPTS / NPRIM;   // 512
static constexpr int   MALL   = NPTS + 5000;    // 13192
static constexpr int   NRES   = BB * NPTS;      // 32768
static constexpr int   NDEC   = BB * SS;        // 2048
static constexpr float EPSBN  = 1e-5f;
static constexpr float ALPHA  = 1.5f;

static constexpr long OUT2_OFF = (long)BB * NPTS * 3;       // 98304
static constexpr long LOSS_OFF = 2L * BB * NPTS * 3;        // 196608
static constexpr long MEAN_OFF = LOSS_OFF + 1;

template <int N>
__device__ __forceinline__ void wait_async_le() {
#if ASYNC_STAGE
#if __has_builtin(__builtin_amdgcn_s_wait_asynccnt)
  __builtin_amdgcn_s_wait_asynccnt(N);
#else
  asm volatile("s_wait_asynccnt %0" :: "i"(N) : "memory");
#endif
#endif
}

// one 16-byte lane copy: global -> LDS
__device__ __forceinline__ void cp_async16(const _Float16* g, _Float16* l) {
#if ASYNC_STAGE
  unsigned loff = (unsigned)(unsigned long long)l;   // low 32b of flat = LDS addr
  asm volatile("global_load_async_to_lds_b128 %0, %1, off"
               :: "v"(loff), "v"(g) : "memory");
#else
  *(v8h*)l = *(const v8h*)g;
#endif
}

// ---------------------------------------------------------------- utilities
__global__ void k_zeroscal(float* dout) {
  if (threadIdx.x < 5) dout[LOSS_OFF + threadIdx.x] = 0.f;
}

// weights f32 -> f16, padded [G][Opad][Cpad], optional source column offset
__global__ void k_convw(const float* __restrict__ src, _Float16* __restrict__ dst,
                        int G, int O, int C, int Opad, int Cpad, int colOff, int srcC) {
  long total = (long)G * Opad * Cpad;
  long i = blockIdx.x * 256L + threadIdx.x;
  if (i >= total) return;
  int c = (int)(i % Cpad);
  long r = i / Cpad;
  int o = (int)(r % Opad);
  int g = (int)(r / Opad);
  float v = 0.f;
  if (o < O && c < C) v = src[((long)g * O + o) * srcC + colOff + c];
  dst[i] = (_Float16)v;
}

// decoder input Y [16][2048][288] f16 (point-major): cols 0-1 rand_grid,
// 2-257 feat, rest 0
__global__ void k_buildY(const float* __restrict__ rg, const float* __restrict__ feat,
                         _Float16* __restrict__ Y) {
  long total = 16L * NDEC * 288;
  long i = blockIdx.x * 256L + threadIdx.x;
  if (i >= total) return;
  int row = (int)(i % 288);          // channel
  long r = i / 288;
  int col = (int)(r % NDEC);         // point
  int p = (int)(r / NDEC);
  int b = col >> 9, s = col & 511;
  float v = 0.f;
  if (row < 2)        v = rg[(((long)p * 4 + b) * 2 + row) * SS + s];
  else if (row < 258) v = feat[b * 256 + (row - 2)];
  Y[i] = (_Float16)v;
}

// ------------------------------------------------------------- WMMA GEMM
// Z[g][m][n] = sum_k A[g][m][k] * B[g][n][k]   (B is point-major [N][Kpad])
// A: f16 [Mpad64][Kpad], B: f16 [N][Kpad], Z: f32 [M][N].
// Block: 8 waves -> 64x128 tile; wave: 2x2 accs (32x32); K-step 32,
// double-buffered async global->LDS staging.
__global__ __launch_bounds__(256)
void k_wmma_gemm(const _Float16* __restrict__ A, long strideAg,
                 const _Float16* __restrict__ Bm, long strideBg,
                 float* __restrict__ Z, long strideZg,
                 int M, int N, int Kpad,
                 const float* __restrict__ colAdd, int NperB, int addStride) {
  __shared__ __align__(16) _Float16 sA[2][64 * 32];
  __shared__ __align__(16) _Float16 sB[2][128 * 32];
  const int g  = blockIdx.z;
  const int m0 = blockIdx.y * 64;
  const int n0 = blockIdx.x * 128;
  const _Float16* Ag = A + (long)g * strideAg;
  const _Float16* Bg = Bm + (long)g * strideBg;
  float* Zg = Z + (long)g * strideZg;
  const int t = threadIdx.x;
  const int lane = t & 31, wave = t >> 5;
  const int wm = wave & 1, wn = wave >> 1;          // 2 x 4 wave grid
  const int lm = lane & 15;
  const int kbA = (lane < 16) ? 0 : 16;             // A-fragment split, contiguous k = 16h + s
  const int kbB = (lane < 16) ? 0 : 16;             // B-fragment split, contiguous k = 16h + s

  const int arow = t >> 2, ach = (t & 3) * 8;       // A stage: 16B / thread
  const int brow = t >> 1, boff = (t & 1) * 16;     // B stage: 32B / thread

  v8f acc00 = {0.f,0.f,0.f,0.f,0.f,0.f,0.f,0.f};
  v8f acc01 = acc00, acc10 = acc00, acc11 = acc00;

  auto stage = [&](int k0, int buf) {
    cp_async16(Ag + (long)(m0 + arow) * Kpad + k0 + ach, &sA[buf][arow * 32 + ach]);
    cp_async16(Bg + (long)(n0 + brow) * Kpad + k0 + boff, &sB[buf][brow * 32 + boff]);
    cp_async16(Bg + (long)(n0 + brow) * Kpad + k0 + boff + 8, &sB[buf][brow * 32 + boff + 8]);
  };

  const int nsteps = Kpad >> 5;
  stage(0, 0);
  for (int s = 0; s < nsteps; ++s) {
    const int cur = s & 1;
    if (s + 1 < nsteps) { stage((s + 1) * 32, cur ^ 1); wait_async_le<3>(); }
    else                { wait_async_le<0>(); }
    __syncthreads();
    const _Float16* cA = sA[cur];
    const _Float16* cB = sB[cur];
    // A fragments for rows wm*32 + {0,16} + lm (halves 0-7: K=kbA.., 8-15: K=kbA+8..)
    v8h a0lo = *(const v8h*)(cA + (wm * 32 + lm) * 32 + kbA);
    v8h a0hi = *(const v8h*)(cA + (wm * 32 + lm) * 32 + kbA + 8);
    v8h a1lo = *(const v8h*)(cA + (wm * 32 + 16 + lm) * 32 + kbA);
    v8h a1hi = *(const v8h*)(cA + (wm * 32 + 16 + lm) * 32 + kbA + 8);
    // B fragments for cols wn*32 + {0,16} + lm (halves h: K = kbB + h)
    v8h b0lo = *(const v8h*)(cB + (wn * 32 + lm) * 32 + kbB);
    v8h b0hi = *(const v8h*)(cB + (wn * 32 + lm) * 32 + kbB + 8);
    v8h b1lo = *(const v8h*)(cB + (wn * 32 + 16 + lm) * 32 + kbB);
    v8h b1hi = *(const v8h*)(cB + (wn * 32 + 16 + lm) * 32 + kbB + 8);
    v16h a0, a1, b0, b1;
#pragma unroll
    for (int i = 0; i < 8; ++i) {
      a0[i] = a0lo[i]; a0[i + 8] = a0hi[i];
      a1[i] = a1lo[i]; a1[i + 8] = a1hi[i];
      b0[i] = b0lo[i]; b0[i + 8] = b0hi[i];
      b1[i] = b1lo[i]; b1[i + 8] = b1hi[i];
    }
    acc00 = __builtin_amdgcn_wmma_f32_16x16x32_f16(false, a0, false, b0, (short)0, acc00, false, false);
    acc01 = __builtin_amdgcn_wmma_f32_16x16x32_f16(false, a0, false, b1, (short)0, acc01, false, false);
    acc10 = __builtin_amdgcn_wmma_f32_16x16x32_f16(false, a1, false, b0, (short)0, acc10, false, false);
    acc11 = __builtin_amdgcn_wmma_f32_16x16x32_f16(false, a1, false, b1, (short)0, acc11, false, false);
    __syncthreads();
  }
  // C/D layout: VGPR v, lanes 0-15 -> M=v, lanes 16-31 -> M=8+v; N = lane%16
  const int mhalf = (lane < 16) ? 0 : 8;
#pragma unroll
  for (int ti = 0; ti < 2; ++ti) {
#pragma unroll
    for (int tj = 0; tj < 2; ++tj) {
      v8f a = ti ? (tj ? acc11 : acc10) : (tj ? acc01 : acc00);
      int mbase = m0 + wm * 32 + ti * 16 + mhalf;
      int nc = n0 + wn * 32 + tj * 16 + lm;
#pragma unroll
      for (int v = 0; v < 8; ++v) {
        int mr = mbase + v;
        if (mr < M) {
          float val = a[v];
          if (colAdd) val += colAdd[(nc / NperB) * addStride + mr];
          Zg[(long)mr * N + nc] = val;
        }
      }
    }
  }
}

// ----------------------------------------------------------------- BN etc.
__global__ __launch_bounds__(256)
void k_bnstats(const float* __restrict__ Z, long strideG, int N,
               float* __restrict__ mean, float* __restrict__ var, int Mreal) {
  int row = blockIdx.x, g = blockIdx.y;
  const float* zr = Z + (long)g * strideG + (long)row * N;
  float s1 = 0.f, s2 = 0.f;
  for (int n = threadIdx.x; n < N; n += 256) { float v = zr[n]; s1 += v; s2 += v * v; }
  __shared__ float a1[256], a2[256];
  a1[threadIdx.x] = s1; a2[threadIdx.x] = s2;
  __syncthreads();
  for (int o = 128; o > 0; o >>= 1) {
    if (threadIdx.x < o) { a1[threadIdx.x] += a1[threadIdx.x + o]; a2[threadIdx.x] += a2[threadIdx.x + o]; }
    __syncthreads();
  }
  if (threadIdx.x == 0) {
    float m = a1[0] / N;
    mean[g * Mreal + row] = m;
    var[g * Mreal + row] = a2[0] / N - m * m;
  }
}

// BN + optional relu, f32 Z [M][N] -> f16 activation point-major [N][MpadOut]
__global__ void k_bnapply(const float* __restrict__ Z, long zStrideG, int N,
                          int Mreal, int MpadOut,
                          const float* __restrict__ mean, const float* __restrict__ var,
                          const float* __restrict__ gam, const float* __restrict__ bet,
                          int gbStride, _Float16* __restrict__ dst, long dStrideG, int relu) {
  long i = blockIdx.x * 256L + threadIdx.x;
  int g = blockIdx.y;
  long total = (long)MpadOut * N;
  if (i >= total) return;
  int row = (int)(i % MpadOut);
  int col = (int)(i / MpadOut);
  float v = 0.f;
  if (row < Mreal) {
    float z = Z[(long)g * zStrideG + (long)row * N + col];
    float m = mean[g * Mreal + row], vv = fmaxf(var[g * Mreal + row], 0.f);
    v = (z - m) * rsqrtf(vv + EPSBN) * gam[g * gbStride + row] + bet[g * gbStride + row];
    if (relu) v = fmaxf(v, 0.f);
  }
  dst[(long)g * dStrideG + (long)col * MpadOut + row] = (_Float16)v;
}

// decoder layer-4 epilogue: add bias, write out1 and xx_full
__global__ void k_decfinish(const float* __restrict__ Z4, const float* __restrict__ db4,
                            float* __restrict__ dout, float* __restrict__ xxf) {
  long i = blockIdx.x * 256L + threadIdx.x;
  if (i >= 16L * 4 * SS) return;
  int s = (int)(i % SS);
  long r = i / SS;
  int b = (int)(r % 4), p = (int)(r / 4);
  int m = p * SS + s;
  for (int o = 0; o < 3; ++o) {
    float v = Z4[((long)p * 64 + o) * NDEC + b * SS + s] + db4[p * 3 + o];
    dout[((long)b * NPTS + m) * 3 + o] = v;                 // out1 [B][N][3]
    xxf[((long)b * 4 + o) * MALL + m] = v;
  }
  xxf[((long)b * 4 + 3) * MALL + m] = 0.f;
}

__global__ void k_fillpart(const float* __restrict__ partial, float* __restrict__ xxf) {
  long i = blockIdx.x * 256L + threadIdx.x;
  if (i >= 4L * 5000) return;
  int j = (int)(i % 5000), b = (int)(i / 5000);
  for (int o = 0; o < 3; ++o)
    xxf[((long)b * 4 + o) * MALL + NPTS + j] = partial[((long)b * 3 + o) * 5000 + j];
  xxf[((long)b * 4 + 3) * MALL + NPTS + j] = 1.f;
}

// ----------------------------------------------------------- Prim's MST
__global__ __launch_bounds__(512)
void k_mst(const float* __restrict__ out1, float* __restrict__ dout) {
  __shared__ float px[512], py[512], pz[512], keymin[512], sval[512], lsv[512];
  __shared__ int intree[512], sidx[512], jsv[512];
  int bp = blockIdx.x, b = bp >> 4, p = bp & 15;
  int t = threadIdx.x;
  const float* base = out1 + ((long)b * NPTS + p * SS) * 3;
  px[t] = base[t * 3]; py[t] = base[t * 3 + 1]; pz[t] = base[t * 3 + 2];
  __syncthreads();
  {
    float dx = px[t] - px[0], dy = py[t] - py[0], dz = pz[t] - pz[0];
    keymin[t] = sqrtf(dx * dx + dy * dy + dz * dz + 1e-12f);
  }
  intree[t] = (t == 0) ? 1 : 0;
  __syncthreads();
  for (int it = 0; it < SS - 1; ++it) {
    sval[t] = intree[t] ? __builtin_inff() : keymin[t];
    sidx[t] = t;
    __syncthreads();
    for (int o = 256; o > 0; o >>= 1) {
      if (t < o) {
        float v2 = sval[t + o]; int i2 = sidx[t + o];
        if (v2 < sval[t] || (v2 == sval[t] && i2 < sidx[t])) { sval[t] = v2; sidx[t] = i2; }
      }
      __syncthreads();
    }
    int j = sidx[0]; float l = sval[0];
    if (t == 0) { jsv[it] = j; lsv[it] = l; intree[j] = 1; }
    __syncthreads();
    float dx = px[t] - px[j], dy = py[t] - py[j], dz = pz[t] - pz[j];
    keymin[t] = fminf(keymin[t], sqrtf(dx * dx + dy * dy + dz * dz + 1e-12f));
    __syncthreads();
  }
  sval[t] = (t < SS - 1) ? lsv[t] : 0.f;
  __syncthreads();
  for (int o = 256; o > 0; o >>= 1) { if (t < o) sval[t] += sval[t + o]; __syncthreads(); }
  float meanl = sval[0] / (float)(SS - 1);
  __syncthreads();
  sval[t] = (t < SS - 1 && lsv[t] > ALPHA * meanl) ? lsv[t] : 0.f;
  __syncthreads();
  for (int o = 256; o > 0; o >>= 1) { if (t < o) sval[t] += sval[t + o]; __syncthreads(); }
  if (t == 0) {
    atomicAdd(dout + LOSS_OFF, sval[0] * (1.f / (float)(BB * NPTS)));
    atomicAdd(dout + MEAN_OFF + b, meanl * (1.f / (float)NPRIM));
  }
}

// --------------------------------------------------- greedy MDS sampling
__global__ __launch_bounds__(1024)
void k_mds(const float* __restrict__ xxf, const float* __restrict__ dout,
           int* __restrict__ idxs) {
  __shared__ float dens[MALL];                 // 52768 B
  __shared__ float sval[1024];
  __shared__ int   sidx[1024];
  int b = blockIdx.x, t = threadIdx.x;
  const float* X = xxf + (long)b * 4 * MALL;
  const float* Yp = X + MALL;
  const float* Zp = X + 2 * MALL;
  float sig = dout[MEAN_OFF + b];
  float inv = 1.f / (2.f * sig * sig);
  for (int i = t; i < MALL; i += 1024) dens[i] = 0.f;
  __syncthreads();
  for (int step = 0; step < NPTS; ++step) {
    float mv = __builtin_inff(); int mi = 0x7fffffff;
    for (int i = t; i < MALL; i += 1024) {
      float d = dens[i];
      if (d < mv) { mv = d; mi = i; }
    }
    sval[t] = mv; sidx[t] = mi;
    __syncthreads();
    for (int o = 512; o > 0; o >>= 1) {
      if (t < o) {
        float v2 = sval[t + o]; int i2 = sidx[t + o];
        if (v2 < sval[t] || (v2 == sval[t] && i2 < sidx[t])) { sval[t] = v2; sidx[t] = i2; }
      }
      __syncthreads();
    }
    int idx = sidx[0];
    if (t == 0) idxs[b * NPTS + step] = idx;
    float qx = X[idx], qy = Yp[idx], qz = Zp[idx];
    for (int i = t; i < MALL; i += 1024) {
      float dx = X[i] - qx, dy = Yp[i] - qy, dz = Zp[i] - qz;
      float nd = dens[i] + __expf(-(dx * dx + dy * dy + dz * dz) * inv);
      dens[i] = (i == idx) ? __builtin_inff() : nd;
    }
    __syncthreads();
  }
}

// gather sampled points -> xxg [B][4][N] f32, A0 point-major [32768][32] f16
__global__ void k_gather(const float* __restrict__ xxf, const int* __restrict__ idxs,
                         float* __restrict__ xxg, _Float16* __restrict__ A0) {
  long i = blockIdx.x * 256L + threadIdx.x;
  if (i >= 4L * NPTS) return;
  int n = (int)(i % NPTS), b = (int)(i / NPTS);
  int idx = idxs[b * NPTS + n];
  _Float16* a0 = A0 + ((long)b * NPTS + n) * 32;
  for (int c = 0; c < 4; ++c) {
    float v = xxf[((long)b * 4 + c) * MALL + idx];
    xxg[((long)b * 4 + c) * NPTS + n] = v;
    a0[c] = (_Float16)v;
  }
  for (int c = 4; c < 32; ++c) a0[c] = (_Float16)0.f;
}

// r3: BN (no relu) fused with per-batch row max
__global__ __launch_bounds__(256)
void k_bnmax(const float* __restrict__ Z, const float* __restrict__ mean,
             const float* __restrict__ var, const float* __restrict__ gam,
             const float* __restrict__ bet, float* __restrict__ gmax, int rowOff) {
  int row = blockIdx.x, b = blockIdx.y;
  const float* zr = Z + (long)row * NRES + b * NPTS;
  float m = mean[row], vv = fmaxf(var[row], 0.f);
  float sc = rsqrtf(vv + EPSBN) * gam[row];
  float bo = bet[row] - m * sc;
  float mx = -__builtin_inff();
  for (int n = threadIdx.x; n < NPTS; n += 256) mx = fmaxf(mx, zr[n] * sc + bo);
  __shared__ float a[256];
  a[threadIdx.x] = mx;
  __syncthreads();
  for (int o = 128; o > 0; o >>= 1) {
    if (threadIdx.x < o) a[threadIdx.x] = fmaxf(a[threadIdx.x], a[threadIdx.x + o]);
    __syncthreads();
  }
  if (threadIdx.x == 0) gmax[b * 1024 + rowOff + row] = a[0];
}

// gvec[b][o] = sum_c rW4[o][c] * gmax[b][c]  (c < 1024)
__global__ __launch_bounds__(256)
void k_gvec(const float* __restrict__ W4, const float* __restrict__ gmax,
            float* __restrict__ gvec) {
  int o = blockIdx.x, b = blockIdx.y;
  float s = 0.f;
  for (int c = threadIdx.x; c < 1024; c += 256) s += W4[(long)o * 1088 + c] * gmax[b * 1024 + c];
  __shared__ float a[256];
  a[threadIdx.x] = s;
  __syncthreads();
  for (int off = 128; off > 0; off >>= 1) {
    if (threadIdx.x < off) a[threadIdx.x] += a[threadIdx.x + off];
    __syncthreads();
  }
  if (threadIdx.x == 0) gvec[b * 512 + o] = a[0];
}

__global__ void k_out2(const float* __restrict__ Z7, const float* __restrict__ rb7,
                       const float* __restrict__ xxg, float* __restrict__ dout) {
  long i = blockIdx.x * 256L + threadIdx.x;
  if (i >= 4L * NPTS) return;
  int n = (int)(i % NPTS), b = (int)(i / NPTS);
  for (int o = 0; o < 3; ++o) {
    float z = Z7[(long)o * NRES + b * NPTS + n] + rb7[o];
    dout[OUT2_OFF + ((long)b * NPTS + n) * 3 + o] =
        xxg[((long)b * 4 + o) * NPTS + n] + tanhf(z);
  }
}

// ======================================================================
extern "C" void kernel_launch(void* const* d_in, const int* in_sizes, int n_in,
                              void* d_out, int out_size, void* d_ws, size_t ws_size,
                              hipStream_t stream) {
  (void)in_sizes; (void)n_in; (void)out_size; (void)ws_size;
  auto in = [&](int i) { return (const float*)d_in[i]; };
  const float* partial = in(0);
  const float* feat    = in(1);
  const float* randg   = in(2);
  const float* dW[4]   = {in(3), in(7), in(11), in(15)};
  const float* dg[3]   = {in(5), in(9), in(13)};
  const float* dbe[3]  = {in(6), in(10), in(14)};
  const float* db4     = in(16);
  const float* rW[7]   = {in(17), in(21), in(25), in(29), in(33), in(37), in(41)};
  const float* rg[6]   = {in(19), in(23), in(27), in(31), in(35), in(39)};
  const float* rbe[6]  = {in(20), in(24), in(28), in(32), in(36), in(40)};
  const float* rb7     = in(42);
  float* dout = (float*)d_out;

  // ---- workspace carve (linear, 256B aligned) ----
  char* w = (char*)d_ws;
  auto alloc = [&](long bytes) -> char* { char* p = w; w += ((bytes + 255) & ~255L); return p; };
  _Float16* WhD1 = (_Float16*)alloc(16L * 320 * 288 * 2);
  _Float16* WhD2 = (_Float16*)alloc(16L * 192 * 288 * 2);
  _Float16* WhD3 = (_Float16*)alloc(16L * 64 * 160 * 2);
  _Float16* WhD4 = (_Float16*)alloc(16L * 64 * 64 * 2);
  _Float16* WhR1 = (_Float16*)alloc(64L * 32 * 2);
  _Float16* WhR2 = (_Float16*)alloc(128L * 64 * 2);
  _Float16* WhR3 = (_Float16*)alloc(1024L * 128 * 2);
  _Float16* WhR4 = (_Float16*)alloc(512L * 64 * 2);
  _Float16* WhR5 = (_Float16*)alloc(256L * 512 * 2);
  _Float16* WhR6 = (_Float16*)alloc(128L * 256 * 2);
  _Float16* WhR7 = (_Float16*)alloc(64L * 128 * 2);
  _Float16* Yh   = (_Float16*)alloc(16L * NDEC * 288 * 2);   // point-major
  _Float16* X1h  = (_Float16*)alloc(16L * NDEC * 288 * 2);
  _Float16* X2h  = (_Float16*)alloc(16L * NDEC * 160 * 2);
  _Float16* X3h  = (_Float16*)alloc(16L * NDEC * 64 * 2);
  float*    Zb   = (float*)alloc(512L * NRES * 4);           // 67MB, reused everywhere
  float*    XXF  = (float*)alloc(4L * 4 * MALL * 4);
  int*      IDX  = (int*)alloc(4L * NPTS * 4);
  float*    XXG  = (float*)alloc(4L * 4 * NPTS * 4);
  _Float16* A0h  = (_Float16*)alloc((long)NRES * 32 * 2);    // point-major
  _Float16* PFh  = (_Float16*)alloc((long)NRES * 64 * 2);
  _Float16* A2h  = (_Float16*)alloc((long)NRES * 128 * 2);
  float*    GMX  = (float*)alloc(4L * 1024 * 4);
  float*    GVC  = (float*)alloc(4L * 512 * 4);
  _Float16* A4h  = (_Float16*)alloc((long)NRES * 512 * 2);
  _Float16* A5h  = (_Float16*)alloc((long)NRES * 256 * 2);
  _Float16* A6h  = (_Float16*)alloc((long)NRES * 128 * 2);
  float*    MEANB = (float*)alloc(8192L * 4);
  float*    VARB  = (float*)alloc(8192L * 4);

  auto g1 = [](long n) { return dim3((unsigned)((n + 255) / 256)); };

  k_zeroscal<<<1, 32, 0, stream>>>(dout);

  // ---- weights -> f16 padded ----
  k_convw<<<g1(16L*320*288),256,0,stream>>>(dW[0], WhD1, 16, 258, 258, 320, 288, 0, 258);
  k_convw<<<g1(16L*192*288),256,0,stream>>>(dW[1], WhD2, 16, 129, 258, 192, 288, 0, 258);
  k_convw<<<g1(16L*64*160), 256,0,stream>>>(dW[2], WhD3, 16, 64, 129, 64, 160, 0, 129);
  k_convw<<<g1(16L*64*64),  256,0,stream>>>(dW[3], WhD4, 16, 3, 64, 64, 64, 0, 64);
  k_convw<<<g1(64L*32),     256,0,stream>>>(rW[0], WhR1, 1, 64, 4, 64, 32, 0, 4);
  k_convw<<<g1(128L*64),    256,0,stream>>>(rW[1], WhR2, 1, 128, 64, 128, 64, 0, 64);
  k_convw<<<g1(1024L*128),  256,0,stream>>>(rW[2], WhR3, 1, 1024, 128, 1024, 128, 0, 128);
  k_convw<<<g1(512L*64),    256,0,stream>>>(rW[3], WhR4, 1, 512, 64, 512, 64, 1024, 1088);
  k_convw<<<g1(256L*512),   256,0,stream>>>(rW[4], WhR5, 1, 256, 512, 256, 512, 0, 512);
  k_convw<<<g1(128L*256),   256,0,stream>>>(rW[5], WhR6, 1, 128, 256, 128, 256, 0, 256);
  k_convw<<<g1(64L*128),    256,0,stream>>>(rW[6], WhR7, 1, 3, 128, 64, 128, 0, 128);

  // ---- decoder (16 primitives as GEMM groups, N = B*S = 2048) ----
  k_buildY<<<g1(16L*NDEC*288), 256, 0, stream>>>(randg, feat, Yh);

  k_wmma_gemm<<<dim3(NDEC/128, 5, 16), 256, 0, stream>>>(
      WhD1, 320L*288, Yh, (long)NDEC*288, Zb, 320L*NDEC, 258, NDEC, 288, nullptr, 1, 0);
  k_bnstats<<<dim3(258,16), 256, 0, stream>>>(Zb, 320L*NDEC, NDEC, MEANB, VARB, 258);
  k_bnapply<<<dim3(288*NDEC/256,16), 256, 0, stream>>>(
      Zb, 320L*NDEC, NDEC, 258, 288, MEANB, VARB, dg[0], dbe[0], 258, X1h, (long)NDEC*288, 1);

  k_wmma_gemm<<<dim3(NDEC/128, 3, 16), 256, 0, stream>>>(
      WhD2, 192L*288, X1h, (long)NDEC*288, Zb, 192L*NDEC, 129, NDEC, 288, nullptr, 1, 0);
  k_bnstats<<<dim3(129,16), 256, 0, stream>>>(Zb, 192L*NDEC, NDEC, MEANB, VARB, 129);
  k_bnapply<<<dim3(160*NDEC/256,16), 256, 0, stream>>>(
      Zb, 192L*NDEC, NDEC, 129, 160, MEANB, VARB, dg[1], dbe[1], 129, X2h, (long)NDEC*160, 1);

  k_wmma_gemm<<<dim3(NDEC/128, 1, 16), 256, 0, stream>>>(
      WhD3, 64L*160, X2h, (long)NDEC*160, Zb, 64L*NDEC, 64, NDEC, 160, nullptr, 1, 0);
  k_bnstats<<<dim3(64,16), 256, 0, stream>>>(Zb, 64L*NDEC, NDEC, MEANB, VARB, 64);
  k_bnapply<<<dim3(64*NDEC/256,16), 256, 0, stream>>>(
      Zb, 64L*NDEC, NDEC, 64, 64, MEANB, VARB, dg[2], dbe[2], 64, X3h, (long)NDEC*64, 1);

  k_wmma_gemm<<<dim3(NDEC/128, 1, 16), 256, 0, stream>>>(
      WhD4, 64L*64, X3h, (long)NDEC*64, Zb, 64L*NDEC, 3, NDEC, 64, nullptr, 1, 0);
  k_decfinish<<<g1(16L*4*SS), 256, 0, stream>>>(Zb, db4, dout, XXF);
  k_fillpart<<<g1(4L*5000), 256, 0, stream>>>(partial, XXF);

  // ---- MST + expansion penalty, then minimum-density sampling ----
  k_mst<<<64, 512, 0, stream>>>(dout, dout);
  k_mds<<<4, 1024, 0, stream>>>(XXF, dout, IDX);
  k_gather<<<g1(4L*NPTS), 256, 0, stream>>>(XXF, IDX, XXG, A0h);

  // ---- PointNetRes (G = 1, N = B*NPTS = 32768) ----
  k_wmma_gemm<<<dim3(NRES/128, 1, 1), 256, 0, stream>>>(
      WhR1, 0, A0h, 0, Zb, 0, 64, NRES, 32, nullptr, 1, 0);
  k_bnstats<<<dim3(64,1), 256, 0, stream>>>(Zb, 0, NRES, MEANB, VARB, 64);
  k_bnapply<<<dim3(64L*NRES/256,1), 256, 0, stream>>>(
      Zb, 0, NRES, 64, 64, MEANB, VARB, rg[0], rbe[0], 64, PFh, 0, 1);

  k_wmma_gemm<<<dim3(NRES/128, 2, 1), 256, 0, stream>>>(
      WhR2, 0, PFh, 0, Zb, 0, 128, NRES, 64, nullptr, 1, 0);
  k_bnstats<<<dim3(128,1), 256, 0, stream>>>(Zb, 0, NRES, MEANB, VARB, 128);
  k_bnapply<<<dim3(128L*NRES/256,1), 256, 0, stream>>>(
      Zb, 0, NRES, 128, 128, MEANB, VARB, rg[1], rbe[1], 128, A2h, 0, 1);

  // r3: 1024 rows in 4 chunks, BN fused with per-batch max (only gmax kept,
  // so row-chunking has no layout hazard)
  for (int c = 0; c < 4; ++c) {
    k_wmma_gemm<<<dim3(NRES/128, 4, 1), 256, 0, stream>>>(
        WhR3 + (long)c*256*128, 0, A2h, 0, Zb, 0, 256, NRES, 128, nullptr, 1, 0);
    k_bnstats<<<dim3(256,1), 256, 0, stream>>>(Zb, 0, NRES, MEANB, VARB, 256);
    k_bnmax<<<dim3(256,4), 256, 0, stream>>>(Zb, MEANB, VARB, rg[2]+c*256, rbe[2]+c*256,
                                             GMX, c*256);
  }
  k_gvec<<<dim3(512,4), 256, 0, stream>>>(rW[3], GMX, GVC);

  // r4: single pass (M=512), K reduced to 64 (pointfeat part), gmax
  // contribution as per-(b,row) column add; output A4h is a consistent
  // point-major [N][512] buffer for r5.
  k_wmma_gemm<<<dim3(NRES/128, 8, 1), 256, 0, stream>>>(
      WhR4, 0, PFh, 0, Zb, 0, 512, NRES, 64, GVC, NPTS, 512);
  k_bnstats<<<dim3(512,1), 256, 0, stream>>>(Zb, 0, NRES, MEANB, VARB, 512);
  k_bnapply<<<dim3(512L*NRES/256,1), 256, 0, stream>>>(
      Zb, 0, NRES, 512, 512, MEANB, VARB, rg[3], rbe[3], 512, A4h, 0, 1);

  k_wmma_gemm<<<dim3(NRES/128, 4, 1), 256, 0, stream>>>(
      WhR5, 0, A4h, 0, Zb, 0, 256, NRES, 512, nullptr, 1, 0);
  k_bnstats<<<dim3(256,1), 256, 0, stream>>>(Zb, 0, NRES, MEANB, VARB, 256);
  k_bnapply<<<dim3(256L*NRES/256,1), 256, 0, stream>>>(
      Zb, 0, NRES, 256, 256, MEANB, VARB, rg[4], rbe[4], 256, A5h, 0, 1);

  k_wmma_gemm<<<dim3(NRES/128, 2, 1), 256, 0, stream>>>(
      WhR6, 0, A5h, 0, Zb, 0, 128, NRES, 256, nullptr, 1, 0);
  k_bnstats<<<dim3(128,1), 256, 0, stream>>>(Zb, 0, NRES, MEANB, VARB, 128);
  k_bnapply<<<dim3(128L*NRES/256,1), 256, 0, stream>>>(
      Zb, 0, NRES, 128, 128, MEANB, VARB, rg[5], rbe[5], 128, A6h, 0, 1);

  k_wmma_gemm<<<dim3(NRES/128, 1, 1), 256, 0, stream>>>(
      WhR7, 0, A6h, 0, Zb, 0, 3, NRES, 128, nullptr, 1, 0);
  k_out2<<<g1(4L*NPTS), 256, 0, stream>>>(Zb, rb7, XXG, dout);
}
